// PolicyNetwork_17549236371841
// MI455X (gfx1250) — compile-verified, matrix operand order checked by reading, not run on hardware
//
#include <hip/hip_runtime.h>
#include <hip/hip_bf16.h>

// ---------------------------------------------------------------------------
// PolicyNetwork forward for gfx1250 (MI455X). All GEMM-shaped math runs on
// v_wmma_f32_16x16x32_f16 (f16 in, f32 accumulate). Attention is flash-style
// so the 8x4096x4096 score tensor is never materialized.
// All GEMMs have K % 32 == 0 (feats/proj_w are zero-padded to K=544), so the
// global->LDS staging path is branchless: float4 loads -> f16 pack -> b128
// LDS stores. V is staged TRANSPOSED in LDS so both flash matmuls load
// operands as contiguous b128 pairs. ~69 MB of d_ws used.
// ---------------------------------------------------------------------------

typedef __attribute__((ext_vector_type(16))) _Float16 v16h;
typedef __attribute__((ext_vector_type(8)))  _Float16 h8;
typedef __attribute__((ext_vector_type(8)))  float    v8f;

#define NE      4096
#define D_MODEL 512
#define NHEAD   8
#define DHEAD   64
#define NLAYER  2
#define KPAD    544      // 513 padded up to a multiple of 32

__device__ __forceinline__ v8f wmma16(v16h a, v16h b, v8f c) {
  // 8 args: (neg_a, A, neg_b, B, c_mod, C, reuse_a, reuse_b)
  return __builtin_amdgcn_wmma_f32_16x16x32_f16(false, a, false, b, (short)0, c,
                                                false, false);
}

__device__ __forceinline__ h8 pack8(float4 a, float4 b) {
  h8 r;
  r[0] = (_Float16)a.x; r[1] = (_Float16)a.y;
  r[2] = (_Float16)a.z; r[3] = (_Float16)a.w;
  r[4] = (_Float16)b.x; r[5] = (_Float16)b.y;
  r[6] = (_Float16)b.z; r[7] = (_Float16)b.w;
  return r;
}

// A-operand (and B-operand for a transposed-stored matrix) loader from a
// row-major [rows][stride] half tile in LDS. ISA 7.12.2 16-bit A 16x32
// layout: lane l holds row (l&15); k(e) = (e&7) + 8*(l>>4) + 16*(e>>3).
// Elements are two contiguous 8-half runs -> 2x ds_load_b128.
__device__ __forceinline__ v16h load_op_rm(const _Float16* t, int row0,
                                           int stride, int k0) {
  int l = threadIdx.x & 31;
  const _Float16* p = t + (size_t)(row0 + (l & 15)) * stride + k0 + 8 * (l >> 4);
  v16h out;
#pragma unroll
  for (int e = 0; e < 16; ++e) out[e] = p[(e & 7) + 16 * (e >> 3)];
  return out;
}

// A-operand straight from global f32 rows (rows = lane&15 off base pointer).
__device__ __forceinline__ v16h load_op_gf32(const float* t, int stride, int k0) {
  int l = threadIdx.x & 31;
  const float* p = t + (size_t)(l & 15) * stride + k0 + 8 * (l >> 4);
  v16h out;
#pragma unroll
  for (int e = 0; e < 16; ++e) out[e] = (_Float16)p[(e & 7) + 16 * (e >> 3)];
  return out;
}

// ---------------------------------------------------------------------------
// Generic GEMM: C[M,N] = act( A[M,K] @ W[N,K]^T + bias [+ res] ), f32 in/out.
// Block tile 128x64, 8 waves of 32x32 (2x2 WMMA tiles), K stepped by 32.
// Requires: K % 32 == 0, lda/ldw % 4 == 0, M % 128 == 0, N % 64 == 0.
// RES/ACT are compile-time so the epilogue has no per-element branches.
// ---------------------------------------------------------------------------
template <bool RES, bool ACT>
__global__ __launch_bounds__(256) void gemm_kernel(
    const float* __restrict__ A, const float* __restrict__ W,
    const float* __restrict__ bias, const float* __restrict__ res,
    float* __restrict__ C, int K, int lda, int ldw, int ldc) {
  __shared__ __align__(16) _Float16 sA[128 * 40];
  __shared__ __align__(16) _Float16 sW[64 * 40];
  int tid = threadIdx.x;
  int wave = tid >> 5, lane = tid & 31;
  int wm = wave >> 1, wn = wave & 1;
  int m0 = blockIdx.y * 128, n0 = blockIdx.x * 64;

  // A staging: thread t covers row (t>>1), 16 cols at (t&1)*16
  const float* aRow = A + (size_t)(m0 + (tid >> 1)) * lda + (tid & 1) * 16;
  _Float16* sAp = sA + (tid >> 1) * 40 + (tid & 1) * 16;
  // W staging: thread t covers row (t>>2), 8 cols at (t&3)*8
  const float* wRow = W + (size_t)(n0 + (tid >> 2)) * ldw + (tid & 3) * 8;
  _Float16* sWp = sW + (tid >> 2) * 40 + (tid & 3) * 8;

  v8f zero = {0.f, 0.f, 0.f, 0.f, 0.f, 0.f, 0.f, 0.f};
  v8f acc[2][2];
  acc[0][0] = zero; acc[0][1] = zero; acc[1][0] = zero; acc[1][1] = zero;

  for (int k0 = 0; k0 < K; k0 += 32) {
    const float4* ap = (const float4*)(aRow + k0);
    float4 a0 = ap[0], a1 = ap[1], a2 = ap[2], a3 = ap[3];
    const float4* wp = (const float4*)(wRow + k0);
    float4 w0 = wp[0], w1 = wp[1];
    if (k0 + 32 < K)   // prefetch next A K-slice (global_prefetch_b8)
      __builtin_prefetch(aRow + k0 + 32, 0, 3);
    *(h8*)(sAp + 0) = pack8(a0, a1);
    *(h8*)(sAp + 8) = pack8(a2, a3);
    *(h8*)(sWp + 0) = pack8(w0, w1);
    __syncthreads();

    v16h va0 = load_op_rm(sA, wm * 32,      40, 0);
    v16h va1 = load_op_rm(sA, wm * 32 + 16, 40, 0);
    v16h vb0 = load_op_rm(sW, wn * 32,      40, 0);
    v16h vb1 = load_op_rm(sW, wn * 32 + 16, 40, 0);
    acc[0][0] = wmma16(va0, vb0, acc[0][0]);
    acc[0][1] = wmma16(va0, vb1, acc[0][1]);
    acc[1][0] = wmma16(va1, vb0, acc[1][0]);
    acc[1][1] = wmma16(va1, vb1, acc[1][1]);
    __syncthreads();
  }

  int hs = lane >> 4, ci = lane & 15;        // C/D layout: m = v + 8*hs, n = ci
  // Each thread touches exactly 2 output columns; hoist the bias loads.
  float bcol0 = bias[n0 + wn * 32 + ci];
  float bcol1 = bias[n0 + wn * 32 + 16 + ci];
#pragma unroll
  for (int mt = 0; mt < 2; ++mt)
#pragma unroll
    for (int nt = 0; nt < 2; ++nt) {
      int col = n0 + wn * 32 + nt * 16 + ci;
      float bc = nt ? bcol1 : bcol0;
#pragma unroll
      for (int vv = 0; vv < 8; ++vv) {
        int row = m0 + wm * 32 + mt * 16 + hs * 8 + vv;
        float val = acc[mt][nt][vv] + bc;
        if constexpr (RES) val += res[(size_t)row * ldc + col];
        if constexpr (ACT) val = fmaxf(val, 0.f);
        C[(size_t)row * ldc + col] = val;
      }
    }
}

// ---------------------------------------------------------------------------
// Flash attention, one (64-query block, head) per workgroup. 4 waves; each
// wave owns 16 query rows. Online softmax in the WMMA C/D register layout.
// Q and K may live in one interleaved E x 1024 buffer (explicit row strides).
// V is staged transposed (sVt[d][k]) so PV operands load as b128 pairs.
// ---------------------------------------------------------------------------
__global__ __launch_bounds__(128) void flash_kernel(
    const float* __restrict__ Q, const float* __restrict__ Kb,
    const float* __restrict__ V, float* __restrict__ O, int ldq, int ldk) {
  __shared__ __align__(16) _Float16 sK [64 * 72];       // [k][d]
  __shared__ __align__(16) _Float16 sVt[64 * 72];       // [d][k] (transposed)
  __shared__ __align__(16) _Float16 sP[4][16 * 72];     // wave-private P tiles
  int tid = threadIdx.x, wave = tid >> 5, lane = tid & 31;
  int h = blockIdx.y;
  int qbase = blockIdx.x * 64 + wave * 16;
  const float* qp = Q + (size_t)qbase * ldq + h * DHEAD;
  v16h qa0 = load_op_gf32(qp, ldq, 0);       // Q operands held in regs all loop
  v16h qa1 = load_op_gf32(qp, ldq, 32);

  v8f zero = {0.f, 0.f, 0.f, 0.f, 0.f, 0.f, 0.f, 0.f};
  v8f oacc[4];
  oacc[0] = zero; oacc[1] = zero; oacc[2] = zero; oacc[3] = zero;
  float rmax[8], rsum[8];
#pragma unroll
  for (int vv = 0; vv < 8; ++vv) { rmax[vv] = -3e30f; rsum[vv] = 0.f; }

  const float scale = 0.125f;                // 1/sqrt(64)
  int hs = lane >> 4, ci = lane & 15;

  // K/V staging: thread t covers k-row (t>>1), 32 d-cols at (t&1)*32
  int srow = tid >> 1, scol = (tid & 1) * 32;
  _Float16* sKp = sK + srow * 72 + scol;

  for (int kb = 0; kb < NE; kb += 64) {
    const float4* kp = (const float4*)(Kb + (size_t)(kb + srow) * ldk + h * DHEAD + scol);
    const float4* vp = (const float4*)(V  + (size_t)(kb + srow) * D_MODEL + h * DHEAD + scol);
#pragma unroll
    for (int i = 0; i < 4; ++i) {            // K: row-major, b128 stores
      float4 k0 = kp[2 * i], k1 = kp[2 * i + 1];
      *(h8*)(sKp + 8 * i) = pack8(k0, k1);
    }
#pragma unroll
    for (int i = 0; i < 8; ++i) {            // V: transposed scatter (b16)
      float4 v4 = vp[i];
      _Float16* c0 = sVt + (scol + 4 * i) * 72 + srow;
      c0[0]       = (_Float16)v4.x;
      c0[72]      = (_Float16)v4.y;
      c0[144]     = (_Float16)v4.z;
      c0[216]     = (_Float16)v4.w;
    }
    __syncthreads();

    v8f s[4];                                // S = Q @ K^T (16x64)
#pragma unroll
    for (int nt = 0; nt < 4; ++nt) {
      s[nt] = wmma16(qa0, load_op_rm(sK, nt * 16, 72, 0),  zero);
      s[nt] = wmma16(qa1, load_op_rm(sK, nt * 16, 72, 32), s[nt]);
    }

#pragma unroll
    for (int vv = 0; vv < 8; ++vv) {         // online softmax per query row
      float tm = fmaxf(fmaxf(s[0][vv], s[1][vv]), fmaxf(s[2][vv], s[3][vv])) * scale;
#pragma unroll
      for (int off = 8; off >= 1; off >>= 1) tm = fmaxf(tm, __shfl_xor(tm, off, 16));
      float nm = fmaxf(rmax[vv], tm);
      float corr = __expf(rmax[vv] - nm);
      rsum[vv] *= corr;
      oacc[0][vv] *= corr; oacc[1][vv] *= corr;
      oacc[2][vv] *= corr; oacc[3][vv] *= corr;
      rmax[vv] = nm;
      float ps = 0.f;
#pragma unroll
      for (int nt = 0; nt < 4; ++nt) {
        float p = __expf(s[nt][vv] * scale - nm);
        ps += p;
        sP[wave][(hs * 8 + vv) * 72 + nt * 16 + ci] = (_Float16)p;
      }
#pragma unroll
      for (int off = 8; off >= 1; off >>= 1) ps += __shfl_xor(ps, off, 16);
      rsum[vv] += ps;
    }

    // Wave-private LDS RAW (same-wave DS ops are in order). O += P @ V.
    v16h pa0 = load_op_rm(sP[wave], 0, 72, 0);
    v16h pa1 = load_op_rm(sP[wave], 0, 72, 32);
#pragma unroll
    for (int dt = 0; dt < 4; ++dt) {
      oacc[dt] = wmma16(pa0, load_op_rm(sVt, dt * 16, 72, 0),  oacc[dt]);
      oacc[dt] = wmma16(pa1, load_op_rm(sVt, dt * 16, 72, 32), oacc[dt]);
    }
    __syncthreads();
  }

#pragma unroll
  for (int dt = 0; dt < 4; ++dt)
#pragma unroll
    for (int vv = 0; vv < 8; ++vv) {
      int row = qbase + hs * 8 + vv;
      O[(size_t)row * D_MODEL + h * DHEAD + dt * 16 + ci] = oacc[dt][vv] / rsum[vv];
    }
}

// ---------------------------------------------------------------------------
// Elementwise / reduction kernels
// ---------------------------------------------------------------------------
__global__ void gather_kernel(const float* __restrict__ emb,
                              const int* __restrict__ ids,
                              const float* __restrict__ ent,
                              float* __restrict__ feats) {
  int e = blockIdx.x;
  int id = ids[e];
  for (int j = threadIdx.x; j < KPAD; j += 256) {
    float v = 0.f;
    if (j < 512)       v = emb[(size_t)id * 512 + j];
    else if (j == 512) v = ent[e];
    feats[(size_t)e * KPAD + j] = v;
  }
}

__global__ void padw_kernel(const float* __restrict__ w, float* __restrict__ wp) {
  int r = blockIdx.x;   // 512 rows
  for (int c = threadIdx.x; c < KPAD; c += 256)
    wp[(size_t)r * KPAD + c] = (c < 513) ? w[(size_t)r * 513 + c] : 0.f;
}

__global__ void rope_kernel(const float* __restrict__ x,
                            const int* __restrict__ ids,
                            const int* __restrict__ NyP,
                            float* __restrict__ xr) {
  int e = blockIdx.x, i = threadIdx.x;      // i in [0,256): half-sel + pair idx
  int Ny = NyP[0];
  int id = ids[e];
  int hsel = i >> 7, j = i & 127;
  float pos = hsel ? (float)(id % Ny) : (float)(id / Ny);
  // theta = 10000^(-j/128)
  float theta = __expf(-(float)j * (9.210340371976184f / 128.f));
  float ang = pos * theta;
  float c = cosf(ang), s = sinf(ang);
  int base = hsel * 256;
  float x1 = x[(size_t)e * 512 + base + j];
  float x2 = x[(size_t)e * 512 + base + 128 + j];
  xr[(size_t)e * 512 + base + j]       = x1 * c - x2 * s;
  xr[(size_t)e * 512 + base + 128 + j] = x1 * s + x2 * c;
}

__global__ __launch_bounds__(256) void ln_kernel(const float* __restrict__ src,
                                                 const float* __restrict__ g,
                                                 const float* __restrict__ b,
                                                 float* __restrict__ dst) {
  int wave = threadIdx.x >> 5, lane = threadIdx.x & 31;
  int row = blockIdx.x * 8 + wave;
  float v[16];
  float sum = 0.f;
#pragma unroll
  for (int j = 0; j < 16; ++j) {
    v[j] = src[(size_t)row * 512 + j * 32 + lane];
    sum += v[j];
  }
#pragma unroll
  for (int off = 16; off >= 1; off >>= 1) sum += __shfl_xor(sum, off, 32);
  float mean = sum * (1.f / 512.f);
  float var = 0.f;
#pragma unroll
  for (int j = 0; j < 16; ++j) { float d = v[j] - mean; var += d * d; }
#pragma unroll
  for (int off = 16; off >= 1; off >>= 1) var += __shfl_xor(var, off, 32);
  float rs = rsqrtf(var * (1.f / 512.f) + 1e-5f);
#pragma unroll
  for (int j = 0; j < 16; ++j) {
    int col = j * 32 + lane;
    dst[(size_t)row * 512 + col] = (v[j] - mean) * rs * g[col] + b[col];
  }
}

__global__ __launch_bounds__(256) void logits_kernel(const float* __restrict__ h2,
                                                     const float* __restrict__ w3,
                                                     const float* __restrict__ b3,
                                                     float* __restrict__ logits) {
  int wave = threadIdx.x >> 5, lane = threadIdx.x & 31;
  int e = blockIdx.x * 8 + wave;
  float sum = 0.f;
#pragma unroll
  for (int j = 0; j < 8; ++j) {
    int c = j * 32 + lane;
    sum += h2[(size_t)e * 256 + c] * w3[c];
  }
#pragma unroll
  for (int off = 16; off >= 1; off >>= 1) sum += __shfl_xor(sum, off, 32);
  if (lane == 0) logits[e] = sum + b3[0];
}

__global__ __launch_bounds__(1024) void softmax_kernel(
    const float* __restrict__ logits, float* __restrict__ probs,
    float* __restrict__ entropy) {
  __shared__ float red[1024];
  int t = threadIdx.x;
  float l[4];
  float m = -3e30f;
#pragma unroll
  for (int i = 0; i < 4; ++i) { l[i] = logits[t + i * 1024]; m = fmaxf(m, l[i]); }
  red[t] = m; __syncthreads();
  for (int s2 = 512; s2 > 0; s2 >>= 1) {
    if (t < s2) red[t] = fmaxf(red[t], red[t + s2]);
    __syncthreads();
  }
  m = red[0]; __syncthreads();
  float sum = 0.f;
#pragma unroll
  for (int i = 0; i < 4; ++i) sum += __expf(l[i] - m);
  red[t] = sum; __syncthreads();
  for (int s2 = 512; s2 > 0; s2 >>= 1) {
    if (t < s2) red[t] += red[t + s2];
    __syncthreads();
  }
  float lse = __logf(red[0]); __syncthreads();
  float ent = 0.f;
#pragma unroll
  for (int i = 0; i < 4; ++i) {
    float lp = l[i] - m - lse;
    float p = __expf(lp);
    probs[t + i * 1024] = p;
    ent -= p * lp;
  }
  red[t] = ent; __syncthreads();
  for (int s2 = 512; s2 > 0; s2 >>= 1) {
    if (t < s2) red[t] += red[t + s2];
    __syncthreads();
  }
  if (t == 0) entropy[0] = red[0];
}

// ---------------------------------------------------------------------------
extern "C" void kernel_launch(void* const* d_in, const int* in_sizes, int n_in,
                              void* d_out, int out_size, void* d_ws, size_t ws_size,
                              hipStream_t stream) {
  const float* node_emb  = (const float*)d_in[0];
  const int*   tile_ids  = (const int*)  d_in[1];
  const int*   NyP       = (const int*)  d_in[2];
  const float* entropies = (const float*)d_in[3];
  const float* proj_w    = (const float*)d_in[4];
  const float* proj_b    = (const float*)d_in[5];
  const float* in_w      = (const float*)d_in[6];
  const float* in_b      = (const float*)d_in[7];
  const float* out_w     = (const float*)d_in[8];
  const float* out_b     = (const float*)d_in[9];
  const float* ff_w1     = (const float*)d_in[10];
  const float* ff_b1     = (const float*)d_in[11];
  const float* ff_w2     = (const float*)d_in[12];
  const float* ff_b2     = (const float*)d_in[13];
  const float* n1_g      = (const float*)d_in[14];
  const float* n1_b      = (const float*)d_in[15];
  const float* n2_g      = (const float*)d_in[16];
  const float* n2_b      = (const float*)d_in[17];
  const float* sh_w1     = (const float*)d_in[18];
  const float* sh_b1     = (const float*)d_in[19];
  const float* sh_w2     = (const float*)d_in[20];
  const float* sh_b2     = (const float*)d_in[21];
  const float* sh_w3     = (const float*)d_in[22];
  const float* sh_b3     = (const float*)d_in[23];

  const size_t E = NE;
  float* ws    = (float*)d_ws;
  float* feats = ws;                   // E*544
  float* wpad  = feats + E * KPAD;     // 512*544 (padded proj_w)
  float* x     = wpad + (size_t)512 * KPAD;  // E*512
  float* xr    = x     + E * 512;      // E*512
  float* qk    = xr    + E * 512;      // E*1024 (interleaved Q|K rows)
  float* vb    = qk    + E * 1024;     // E*512
  float* attn  = vb    + E * 512;      // E*512
  float* x2    = attn  + E * 512;      // E*512   (total ~69 MB)
  float* ffh   = qk;                   // reuse qk region: E*1024
  float* h1    = vb;                   // reuse: E*256
  float* h2    = attn;                 // reuse: E*256

  float* probs  = (float*)d_out;       // [E]
  float* entOut = probs + E;           // [1]
  float* logits = entOut + 1;          // [E]

  // feats = [node_emb[ids], entropy, 0-pad] ; wpad = [proj_w, 0-pad]
  gather_kernel<<<NE, 256, 0, stream>>>(node_emb, tile_ids, entropies, feats);
  padw_kernel<<<512, 256, 0, stream>>>(proj_w, wpad);
  gemm_kernel<false, false><<<dim3(512 / 64, NE / 128), 256, 0, stream>>>(
      feats, wpad, proj_b, nullptr, x, KPAD, KPAD, KPAD, 512);

  for (int l = 0; l < NLAYER; ++l) {
    rope_kernel<<<NE, 256, 0, stream>>>(x, tile_ids, NyP, xr);
    const float* wqk = in_w + (size_t)l * 3 * 512 * 512;  // wq||wk rows 0..1023
    const float* wv  = wqk + (size_t)1024 * 512;
    const float* bqk = in_b + (size_t)l * 1536;
    // Fused Q|K projection: N = 1024, output row-interleaved in qk.
    gemm_kernel<false, false><<<dim3(16, 32), 256, 0, stream>>>(
        xr, wqk, bqk, nullptr, qk, 512, 512, 512, 1024);
    gemm_kernel<false, false><<<dim3(8, 32), 256, 0, stream>>>(
        x, wv, bqk + 1024, nullptr, vb, 512, 512, 512, 512);
    flash_kernel<<<dim3(NE / 64, NHEAD), 128, 0, stream>>>(
        qk, qk + 512, vb, attn, 1024, 1024);
    gemm_kernel<true, false><<<dim3(8, 32), 256, 0, stream>>>(
        attn, out_w + (size_t)l * 512 * 512, out_b + l * 512, x, x2,
        512, 512, 512, 512);
    ln_kernel<<<NE / 8, 256, 0, stream>>>(x2, n1_g + l * 512, n1_b + l * 512, x);
    gemm_kernel<false, true><<<dim3(16, 32), 256, 0, stream>>>(
        x, ff_w1 + (size_t)l * 1024 * 512, ff_b1 + l * 1024, nullptr, ffh,
        512, 512, 512, 1024);
    gemm_kernel<true, false><<<dim3(8, 32), 256, 0, stream>>>(
        ffh, ff_w2 + (size_t)l * 512 * 1024, ff_b2 + l * 512, x, x2,
        1024, 1024, 1024, 512);
    ln_kernel<<<NE / 8, 256, 0, stream>>>(x2, n2_g + l * 512, n2_b + l * 512, x);
  }

  gemm_kernel<false, true><<<dim3(4, 32), 256, 0, stream>>>(
      x, sh_w1, sh_b1, nullptr, h1, 512, 512, 512, 256);
  gemm_kernel<false, true><<<dim3(4, 32), 256, 0, stream>>>(
      h1, sh_w2, sh_b2, nullptr, h2, 256, 256, 256, 256);
  logits_kernel<<<NE / 8, 256, 0, stream>>>(h2, sh_w3, sh_b3, logits);
  softmax_kernel<<<1, 1024, 0, stream>>>(logits, probs, entOut);
}
